// NN_action_value_85263690760202
// MI455X (gfx1250) — matrix operand order repaired and checked by reading, not compiled, not run on hardware
//
#include <hip/hip_runtime.h>

typedef __attribute__((ext_vector_type(16))) _Float16 v16h;
typedef __attribute__((ext_vector_type(8)))  float    v8f;

#define B_G   1024
#define NPG   64
#define DEG   16
#define EPG   (NPG*DEG)          // 1024 edges per graph
#define NEPG  256
#define NTOT  (B_G*NPG)          // 65536
#define ETOT  (NTOT*DEG)
#define EMB   64
#define HID   128
#define NOUTD 101
#define NPAD  112                // 101 padded to 7 n-tiles
#define ROW   (1 + NPG*NOUTD + NEPG)   // 6721

union Frag16 { v16h v; unsigned u[8]; _Float16 e[16]; };

__device__ inline v8f wmma_f16(v16h a, v16h b, v8f c) {
  return __builtin_amdgcn_wmma_f32_16x16x32_f16(false, a, false, b, (short)0, c,
                                                false, false);
}

// ---- CDNA5 async global->LDS (GV mode, per-lane 16B), tracked by ASYNCcnt ----
__device__ inline void async_ld_b128(unsigned lds_off, const void* g) {
  asm volatile("global_load_async_to_lds_b128 %0, %1, off"
               :: "v"(lds_off), "v"(g) : "memory");
}
__device__ inline void async_wait0() {
  asm volatile("s_wait_asynccnt 0x0" ::: "memory");
}
// stage a 16KB (64x128 f16) tile: 256 threads x 4 x 16B
__device__ inline void stage_tile_async(void* lds, const void* g, int tid) {
  const unsigned base = (unsigned)(size_t)lds;      // low 32 bits = LDS offset
  const char* gp = (const char*)g;
#pragma unroll
  for (int j = 0; j < 4; ++j) {
    const unsigned off = (unsigned)(tid + j*256) * 16u;
    async_ld_b128(base + off, gp + off);
  }
  async_wait0();
}

// B fragment from pre-transposed f16 weights Wt[N][K] (global, L2-resident).
// Layout: lane holds col n0+(lane&15); half h -> k = kb+(h&7)+8*(lane>>4)+16*(h>>3).
__device__ inline v16h load_bfrag_t(const _Float16* __restrict__ Wt, int ldk,
                                    int kb, int n0) {
  const int lane = threadIdx.x & 31;
  const int lh   = lane >> 4;
  const unsigned* row = (const unsigned*)(Wt + (n0 + (lane & 15))*ldk);
  Frag16 b;
#pragma unroll
  for (int p = 0; p < 8; ++p) {
    int k = kb + 2*(p & 3) + 8*lh + 16*(p >> 2);
    b.u[p] = row[k >> 1];
  }
  return b.v;
}

// A fragment: rows m0..m0+15 x cols kb..kb+31 of LDS f16 row-major, stride lda.
// Layout: lane holds row m0+(lane&15); half h -> k = kb+(h&7)+8*(lane>>4)+16*(h>>3).
__device__ inline v16h load_afrag_lds(const _Float16* A, int lda, int m0, int kb) {
  const int lane = threadIdx.x & 31;
  const int lh   = lane >> 4;
  const int m    = m0 + (lane & 15);
  const unsigned* row = (const unsigned*)(A + m*lda);
  Frag16 a;
#pragma unroll
  for (int p = 0; p < 8; ++p) {
    int k = kb + 2*(p & 3) + 8*lh + 16*(p >> 2);
    a.u[p] = row[k >> 1];
  }
  return a.v;
}

// ---------------- precompute ea[t] = ((edge_tab[t] @ We) @ ae) per layer -------
__global__ void __launch_bounds__(256)
prep_kernel(const float* __restrict__ edge_tab,
            const float* __restrict__ we1, const float* __restrict__ ae1,
            const float* __restrict__ we2, const float* __restrict__ ae2,
            float* ea1, float* ea2) {
  __shared__ float part[1024];
  const int tid = threadIdx.x;
  for (int w = tid; w < 1024; w += 256) {
    const int layer = w >> 9;
    const int t = (w >> 7) & 3;
    const int j = w & 127;
    const float* we = layer ? we2 : we1;
    const float* ae = layer ? ae2 : ae1;
    float he = 0.f;
    for (int i = 0; i < EMB; ++i) he += edge_tab[t*EMB + i] * we[i*HID + j];
    part[w] = he * ae[j];
  }
  __syncthreads();
  if (tid < 8) {
    const int layer = tid >> 2, t = tid & 3;
    float s = 0.f;
    for (int c = 0; c < 128; ++c) s += part[layer*512 + t*128 + c];
    if (layer) ea2[t] = s; else ea1[t] = s;
  }
}

// ---------------- transpose+convert all weight matrices to f16 [N][K] ----------
__global__ void __launch_bounds__(256)
pack_kernel(const float* __restrict__ c1_w, const float* __restrict__ c2_w,
            const float* __restrict__ n_w1, const float* __restrict__ n_w2,
            const float* __restrict__ e_w1,
            _Float16* wt1, _Float16* wt2, _Float16* wtn1, _Float16* wtn2,
            _Float16* wte1) {
  const int i = blockIdx.x*256 + threadIdx.x;     // up to 128*128
  if (i < 128*64)  wt1[i] = (_Float16)c1_w[(i & 63)*HID + (i >> 6)];
  if (i < 128*128) {
    const int n = i >> 7, k = i & 127;
    wt2[i]  = (_Float16)c2_w[k*HID + n];
    wtn1[i] = (_Float16)n_w1[k*HID + n];
    wte1[i] = (_Float16)e_w1[k*HID + n];
  }
  if (i < NPAD*128) {
    const int n = i >> 7, k = i & 127;
    wtn2[i] = (_Float16)(n < NOUTD ? n_w2[k*NOUTD + n] : 0.f);
  }
}

// ---------------- fused per-graph GATConv --------------------------------------
// one workgroup (8 waves) per graph; everything resident in LDS.
template <int CIN, bool GATHER, bool RELU>
__global__ void __launch_bounds__(256)
gat_kernel(const float* __restrict__ node_tab, const _Float16* __restrict__ hin_g,
           const int* __restrict__ xcat,
           const int* __restrict__ src_g, const int* __restrict__ dst_g,
           const int* __restrict__ attr_g,
           const _Float16* __restrict__ Wt,      // f16 [HID][CIN]
           const float* __restrict__ a_s, const float* __restrict__ a_d,
           const float* __restrict__ bias, const float* __restrict__ ea,
           _Float16* __restrict__ hout) {
  __shared__ __align__(16) char smem[65536];
  _Float16* hin = (_Float16*)smem;                  // [64][CIN]   (<=16KB)
  _Float16* hw  = (_Float16*)(smem + 16384);        // [64][128]   16KB f16
  float*    acc = (float*)(smem + 32768);           // [64][128]   32KB f32
  // overlays on the (dead after phase 2) hin region:
  float*    alpha = (float*)smem;                   // [1024]
  int*      sd    = (int*)(smem + 4096);            // [1024] packed src|dst<<8
  float*    as_l  = (float*)(smem + 8192);          // [64]
  float*    ad_l  = (float*)(smem + 8192 + 256);    // [64]
  unsigned* mkey  = (unsigned*)(smem + 8192 + 512); // [64]
  float*    den   = (float*)(smem + 8192 + 768);    // [64]

  const int tid = threadIdx.x;
  const int g   = blockIdx.x;
  const int nb  = g * NPG;
  const int eb  = g * EPG;
  const int lane = tid & 31, lh = lane >> 4, wave = tid >> 5;

  // phase 1: stage input features (f16) into LDS
  if (GATHER) {
    for (int i = tid; i < NPG*CIN; i += 256) {
      int r = i / CIN, c = i % CIN;
      hin[i] = (_Float16)node_tab[xcat[nb + r]*CIN + c];
    }
  } else {
    stage_tile_async(hin, hin_g + (size_t)nb*CIN, tid);   // async DMA, 16KB
  }
  __syncthreads();

  // phase 2: hw = hin @ W   (64 x CIN @ CIN x 128), WMMA f16, B in registers
  {
    const int n0 = wave * 16;
    v16h bfr[CIN/32];
#pragma unroll
    for (int kk = 0; kk < CIN/32; ++kk)
      bfr[kk] = load_bfrag_t(Wt, CIN, kk*32, n0);
#pragma unroll
    for (int mt = 0; mt < 4; ++mt) {
      v8f d = {};
#pragma unroll
      for (int kk = 0; kk < CIN/32; ++kk)
        d = wmma_f16(load_afrag_lds(hin, CIN, mt*16, kk*32), bfr[kk], d);
#pragma unroll
      for (int v = 0; v < 8; ++v)
        hw[(mt*16 + v + 8*lh)*HID + n0 + (lane & 15)] = (_Float16)d[v];
    }
  }
  __syncthreads();

  // phase 3: attention projections as_[m]=hw[m]@a_s, ad_[m]=hw[m]@a_d
  {
    const int m = tid >> 2, q = tid & 3;
    float ps = 0.f, pd = 0.f;
    for (int c = q*32; c < q*32 + 32; ++c) {
      float h = (float)hw[m*HID + c];
      ps += h * a_s[c];
      pd += h * a_d[c];
    }
    ps += __shfl_xor(ps, 1, 32); ps += __shfl_xor(ps, 2, 32);
    pd += __shfl_xor(pd, 1, 32); pd += __shfl_xor(pd, 2, 32);
    if (q == 0) { as_l[m] = ps; ad_l[m] = pd; }
    if (tid < NPG) { mkey[tid] = 0u; den[tid] = 0.f; }
  }
  __syncthreads();

  // phase 4: edge logits + segment max over dst (LDS atomic on ordered uint key)
  float lg[4]; int edl[4];
#pragma unroll
  for (int j = 0; j < 4; ++j) {
    const int le = j*256 + tid;
    const int e  = eb + le;
    const int sl = src_g[e] - nb;
    const int dl = dst_g[e] - nb;
    float l = as_l[sl] + ad_l[dl] + ea[attr_g[e]];
    l = l > 0.f ? l : 0.2f*l;                       // leaky_relu(0.2)
    lg[j] = l; edl[j] = dl;
    sd[le] = sl | (dl << 8);
    unsigned s = __float_as_uint(l);
    unsigned key = (s & 0x80000000u) ? ~s : (s | 0x80000000u);
    atomicMax(&mkey[dl], key);
  }
  __syncthreads();

  // phase 5: exp(logit - max) and denominator
#pragma unroll
  for (int j = 0; j < 4; ++j) {
    const unsigned k = mkey[edl[j]];
    const unsigned s = (k & 0x80000000u) ? (k & 0x7fffffffu) : ~k;
    const float ex = __expf(lg[j] - __uint_as_float(s));
    lg[j] = ex;
    atomicAdd(&den[edl[j]], ex);
  }
  __syncthreads();

  // phase 6: alpha, zero acc, scatter hw[src]*alpha into acc[dst] (ds_add_f32)
#pragma unroll
  for (int j = 0; j < 4; ++j)
    alpha[j*256 + tid] = lg[j] / (den[edl[j]] + 1e-16f);
  for (int i = tid; i < NPG*HID; i += 256) acc[i] = 0.f;
  __syncthreads();
  {
    const int c  = tid & 127;
    const int eo = tid >> 7;                        // 2 edges x 128 chans / iter
    for (int e0 = 0; e0 < EPG; e0 += 2) {
      const int le = e0 + eo;
      const int s  = sd[le];
      atomicAdd(&acc[(s >> 8)*HID + c], alpha[le] * (float)hw[(s & 255)*HID + c]);
    }
  }
  __syncthreads();

  // phase 7: + bias (+relu), store f16 (possibly in place over hin_g)
  for (int i = tid; i < NPG*HID; i += 256) {
    float v = acc[i] + bias[i & (HID-1)];
    if (RELU) v = fmaxf(v, 0.f);
    hout[(size_t)nb*HID + i] = (_Float16)v;
  }
}

// ---------------- graph readout + stop head ------------------------------------
__global__ void __launch_bounds__(128)
stop_kernel(const _Float16* __restrict__ h2, const float* __restrict__ w1,
            const float* __restrict__ b1, const float* __restrict__ w2,
            const float* __restrict__ b2, float* __restrict__ out) {
  __shared__ float gl[HID];
  __shared__ float red[HID];
  const int j = threadIdx.x, g = blockIdx.x, nb = g*NPG;
  float s = 0.f;
  for (int r = 0; r < NPG; ++r) s += (float)h2[((size_t)nb + r)*HID + j];
  gl[j] = s;
  __syncthreads();
  float z = b1[j];
  for (int k = 0; k < HID; ++k) z += gl[k] * w1[k*HID + j];
  red[j] = fmaxf(z, 0.f) * w2[j];
  __syncthreads();
  for (int off = 64; off > 0; off >>= 1) {
    if (j < off) red[j] += red[j + off];
    __syncthreads();
  }
  if (j == 0) out[(size_t)g*ROW] = red[0] + b2[0];
}

// ---------------- per-node addnode MLP (WMMA) ----------------------------------
__global__ void __launch_bounds__(256)
addnode_kernel(const _Float16* __restrict__ h2,
               const _Float16* __restrict__ wt1,   // f16 [128][128]
               const float* __restrict__ b1,
               const _Float16* __restrict__ wt2,   // f16 [112][128] (padded)
               const float* __restrict__ b2, float* __restrict__ out) {
  __shared__ __align__(16) _Float16 a_t[NPG*HID];   // 16KB
  __shared__ __align__(16) _Float16 z_t[NPG*HID];   // 16KB
  const int tid = threadIdx.x, g = blockIdx.x, nb = g*NPG;
  const int lane = tid & 31, lh = lane >> 4, wave = tid >> 5;
  stage_tile_async(a_t, h2 + (size_t)nb*HID, tid);
  __syncthreads();
  {  // z = relu(a @ w1 + b1)
    const int n0 = wave * 16;
    v16h bfr[4];
#pragma unroll
    for (int kk = 0; kk < 4; ++kk) bfr[kk] = load_bfrag_t(wt1, HID, kk*32, n0);
#pragma unroll
    for (int mt = 0; mt < 4; ++mt) {
      v8f d = {};
#pragma unroll
      for (int kk = 0; kk < 4; ++kk)
        d = wmma_f16(load_afrag_lds(a_t, HID, mt*16, kk*32), bfr[kk], d);
      const int n = n0 + (lane & 15);
      const float bb = b1[n];
#pragma unroll
      for (int v = 0; v < 8; ++v)
        z_t[(mt*16 + v + 8*lh)*HID + n] = (_Float16)fmaxf(d[v] + bb, 0.f);
    }
  }
  __syncthreads();
  // out = z @ w2 + b2   (101 cols -> 7 n-tiles, zero-padded weights)
  for (int t = wave; t < 28; t += 8) {
    const int mt = t & 3, nt = t >> 2;
    const int n0 = nt * 16;
    v16h bfr[4];
#pragma unroll
    for (int kk = 0; kk < 4; ++kk) bfr[kk] = load_bfrag_t(wt2, HID, kk*32, n0);
    v8f d = {};
#pragma unroll
    for (int kk = 0; kk < 4; ++kk)
      d = wmma_f16(load_afrag_lds(z_t, HID, mt*16, kk*32), bfr[kk], d);
    const int n = n0 + (lane & 15);
    if (n < NOUTD) {
      const float bb = b2[n];
#pragma unroll
      for (int v = 0; v < 8; ++v) {
        const int l = mt*16 + v + 8*lh;             // local node id == row
        out[(size_t)g*ROW + 1 + l*NOUTD + n] = d[v] + bb;
      }
    }
  }
}

// ---------------- per-pair addedge MLP (WMMA) ----------------------------------
__global__ void __launch_bounds__(256)
addedge_kernel(const _Float16* __restrict__ h2, const int* __restrict__ ne,
               const _Float16* __restrict__ wt1,  // f16 [128][128]
               const float* __restrict__ b1, const float* __restrict__ w2,
               const float* __restrict__ b2, float* __restrict__ out) {
  __shared__ __align__(16) _Float16 h_t[NPG*HID];   // 16KB
  __shared__ __align__(16) _Float16 p_t[64*HID];    // 16KB
  __shared__ __align__(16) _Float16 z_t[64*HID];    // 16KB
  const int tid = threadIdx.x, g = blockIdx.x, nb = g*NPG;
  const int lane = tid & 31, lh = lane >> 4, wave = tid >> 5;
  const int n0 = wave * 16;
  stage_tile_async(h_t, h2 + (size_t)nb*HID, tid);
  v16h bfr[4];                                      // e_w1 slab, reused all chunks
#pragma unroll
  for (int kk = 0; kk < 4; ++kk) bfr[kk] = load_bfrag_t(wt1, HID, kk*32, n0);
  __syncthreads();
  for (int chunk = 0; chunk < 4; ++chunk) {
    for (int i = tid; i < 64*HID; i += 256) {       // pair = h[i] + h[j]
      const int p = i >> 7, c = i & 127;
      const int pe = (g*NEPG + chunk*64 + p) * 2;
      const int ii = ne[pe] - nb, jj = ne[pe + 1] - nb;
      p_t[i] = (_Float16)((float)h_t[ii*HID + c] + (float)h_t[jj*HID + c]);
    }
    __syncthreads();
#pragma unroll
    for (int mt = 0; mt < 4; ++mt) {                // z = relu(p @ w1 + b1)
      v8f d = {};
#pragma unroll
      for (int kk = 0; kk < 4; ++kk)
        d = wmma_f16(load_afrag_lds(p_t, HID, mt*16, kk*32), bfr[kk], d);
      const int n = n0 + (lane & 15);
      const float bb = b1[n];
#pragma unroll
      for (int v = 0; v < 8; ++v)
        z_t[(mt*16 + v + 8*lh)*HID + n] = (_Float16)fmaxf(d[v] + bb, 0.f);
    }
    __syncthreads();
    {                                               // val = z @ w2 + b2
      const int p = tid >> 2, q = tid & 3;
      float s = 0.f;
      for (int c = q*32; c < q*32 + 32; ++c) s += (float)z_t[p*HID + c] * w2[c];
      s += __shfl_xor(s, 1, 32);
      s += __shfl_xor(s, 2, 32);
      if (q == 0)
        out[(size_t)g*ROW + 1 + NPG*NOUTD + chunk*64 + p] = s + b2[0];
    }
    __syncthreads();
  }
}

extern "C" void kernel_launch(void* const* d_in, const int* in_sizes, int n_in,
                              void* d_out, int out_size, void* d_ws, size_t ws_size,
                              hipStream_t stream) {
  const float* node_tab = (const float*)d_in[0];
  const float* edge_tab = (const float*)d_in[1];
  const float* c1_w  = (const float*)d_in[2];
  const float* c1_we = (const float*)d_in[3];
  const float* c1_as = (const float*)d_in[4];
  const float* c1_ad = (const float*)d_in[5];
  const float* c1_ae = (const float*)d_in[6];
  const float* c1_b  = (const float*)d_in[7];
  const float* c2_w  = (const float*)d_in[8];
  const float* c2_we = (const float*)d_in[9];
  const float* c2_as = (const float*)d_in[10];
  const float* c2_ad = (const float*)d_in[11];
  const float* c2_ae = (const float*)d_in[12];
  const float* c2_b  = (const float*)d_in[13];
  const float* s_w1 = (const float*)d_in[14];
  const float* s_b1 = (const float*)d_in[15];
  const float* s_w2 = (const float*)d_in[16];
  const float* s_b2 = (const float*)d_in[17];
  const float* n_b1 = (const float*)d_in[19];
  const float* n_b2 = (const float*)d_in[21];
  const float* e_b1 = (const float*)d_in[23];
  const float* e_w2 = (const float*)d_in[24];
  const float* e_b2 = (const float*)d_in[25];
  const int* x          = (const int*)d_in[26];
  const int* edge_index = (const int*)d_in[27];
  const int* edge_attr  = (const int*)d_in[28];
  const int* non_edges  = (const int*)d_in[29];

  char* ws = (char*)d_ws;
  float*    ea1 = (float*)ws;
  float*    ea2 = (float*)(ws + 16);
  _Float16* h   = (_Float16*)(ws + 256);            // N x HID f16 (16MB), in-place
  char* wtb = ws + 256 + (size_t)NTOT*HID*2;
  _Float16* wt1  = (_Float16*)(wtb);                          // 128x64
  _Float16* wt2  = (_Float16*)(wtb + 16384);                  // 128x128
  _Float16* wtn1 = (_Float16*)(wtb + 16384 + 32768);          // 128x128
  _Float16* wtn2 = (_Float16*)(wtb + 16384 + 2*32768);        // 112x128
  _Float16* wte1 = (_Float16*)(wtb + 16384 + 2*32768 + 28672);// 128x128
  float* out = (float*)d_out;
  const int* src = edge_index;
  const int* dst = edge_index + ETOT;

  prep_kernel<<<1, 256, 0, stream>>>(edge_tab, c1_we, c1_ae, c2_we, c2_ae, ea1, ea2);
  pack_kernel<<<64, 256, 0, stream>>>((const float*)d_in[2], (const float*)d_in[8],
                                      (const float*)d_in[18], (const float*)d_in[20],
                                      (const float*)d_in[22],
                                      wt1, wt2, wtn1, wtn2, wte1);
  gat_kernel<EMB, true, true><<<B_G, 256, 0, stream>>>(
      node_tab, nullptr, x, src, dst, edge_attr, wt1, c1_as, c1_ad, c1_b, ea1, h);
  gat_kernel<HID, false, false><<<B_G, 256, 0, stream>>>(
      nullptr, h, nullptr, src, dst, edge_attr, wt2, c2_as, c2_ad, c2_b, ea2, h);
  stop_kernel<<<B_G, 128, 0, stream>>>(h, s_w1, s_b1, s_w2, s_b2, out);
  addnode_kernel<<<B_G, 256, 0, stream>>>(h, wtn1, n_b1, wtn2, n_b2, out);
  addedge_kernel<<<B_G, 256, 0, stream>>>(h, non_edges, wte1, e_b1, e_w2, e_b2, out);
  (void)c1_w; (void)c2_w; (void)ws_size; (void)in_sizes; (void)n_in; (void)out_size;
}
